// CrossModalAttention_78640851190177
// MI455X (gfx1250) — matrix operand order repaired and checked by reading, not compiled, not run on hardware
//
#include <hip/hip_runtime.h>

// ---------------------------------------------------------------------------
// CDNA5 (gfx1250) cross-modal attention, WMMA f32_16x16x32_f16 everywhere,
// with GLOBAL_LOAD_ASYNC_TO_LDS_B128 double-buffered staging (ASYNCcnt).
// Pipeline:
//   1) transpose+cvt weights fp32[K][N] -> f16 Wt[N][K]
//   2) GEMM (WMMA): Q = query@Wq+bq  -> head-major f16 [BH][S][64]
//                   K = key  @Wk+bk  -> head-major f16 [BH][S][64]
//                   V = value@Wv+bv  -> transposed  f16 [BH][64][S]
//   3) flash attention (WMMA, transposed-score formulation) -> f16 [B*S][1024]
//   4) GEMM (WMMA): out = attn@Wo+bo -> fp32 d_out
// ---------------------------------------------------------------------------

typedef _Float16 h16v __attribute__((ext_vector_type(16)));
typedef _Float16 h8v  __attribute__((ext_vector_type(8)));
typedef float    f8v  __attribute__((ext_vector_type(8)));
typedef float    f4v  __attribute__((ext_vector_type(4)));

#define WMMA_F16(A, B, C) \
  __builtin_amdgcn_wmma_f32_16x16x32_f16(false, (A), false, (B), (short)0, (C), false, false)

static __device__ __forceinline__ h16v cat8(h8v lo, h8v hi) {
  return __builtin_shufflevector(lo, hi, 0,1,2,3,4,5,6,7,8,9,10,11,12,13,14,15);
}
static __device__ __forceinline__ f8v f8zero() {
  f8v z = {0.f,0.f,0.f,0.f,0.f,0.f,0.f,0.f};
  return z;
}

// Async global->LDS 16-byte copy (per active lane). LDS operand is the byte
// offset within the wave's LDS allocation (low 32 bits of the generic ptr).
static __device__ __forceinline__ void async_g2l_b128(const void* gptr, void* lptr) {
  unsigned      lds = (unsigned)(size_t)lptr;
  unsigned long long ga = (unsigned long long)(size_t)gptr;
  asm volatile("global_load_async_to_lds_b128 %0, %1, off"
               :: "v"(lds), "v"(ga) : "memory");
}

template <int N>
static __device__ __forceinline__ void wait_async() {
#if __has_builtin(__builtin_amdgcn_s_wait_asynccnt)
  __builtin_amdgcn_s_wait_asynccnt((short)N);
#else
  asm volatile("s_wait_asynccnt %0" :: "i"(N) : "memory");
#endif
}

// --------------------------------------------------------------------------
// Weight transpose + fp32->f16:  Wt[n*K + k] = W[k*N + n]
// --------------------------------------------------------------------------
__global__ __launch_bounds__(256) void wcvt_transpose(
    const float* __restrict__ W, _Float16* __restrict__ Wt, int K, int N, int total) {
  int idx = blockIdx.x * 256 + threadIdx.x;
  if (idx >= total) return;
  int n = idx / K;
  int k = idx - n * K;
  Wt[idx] = (_Float16)W[(size_t)k * N + n];
}

// --------------------------------------------------------------------------
// WMMA GEMM with bias: out = A[M][Kd] @ Wt^T + bias   (Wt stored [N][Kd] f16)
// 256 threads, 128x128 block tile, k-step 32, wave tile 32x64 (2x4 frags).
// f16 operand tiles staged via async global->LDS; fp32 A converted in-regs.
// --------------------------------------------------------------------------
enum { OUT_HEAD_ROW = 0, OUT_HEAD_TRANS = 1, OUT_F32_ROW = 2 };

template <typename AT, int OM>
__global__ __launch_bounds__(256) void gemm_wmma_bias(
    const AT* __restrict__ A, const _Float16* __restrict__ Wt,
    const float* __restrict__ bias, void* __restrict__ outp,
    int N, int Kd, int S) {
  __shared__ _Float16 As[128][32];
  __shared__ _Float16 Bs[128][32];

  const int tid  = threadIdx.x;
  const int lane = tid & 31;
  const int w    = tid >> 5;
  const int wm   = w & 3;        // 4 waves along M (32 rows each)
  const int wn   = w >> 2;       // 2 waves along N (64 cols each)
  const int l16  = lane & 15;
  const int hf   = lane >> 4;    // lane half
  const int m0   = blockIdx.y * 128;
  const int n0   = blockIdx.x * 128;

  f8v acc[2][4];
#pragma unroll
  for (int i = 0; i < 2; ++i)
#pragma unroll
    for (int j = 0; j < 4; ++j) acc[i][j] = f8zero();

  for (int k0 = 0; k0 < Kd; k0 += 32) {
    // ---- stage tiles: 512 chunks of 8 halves each for A and B ----
#pragma unroll
    for (int it = 0; it < 2; ++it) {
      int c   = tid + it * 256;      // 0..511
      int row = c >> 2;              // 0..127
      int cc  = (c & 3) * 8;         // 0,8,16,24
      if (sizeof(AT) == 4) {         // fp32 source: convert on the fly (VALU)
        const float* src = (const float*)A + (size_t)(m0 + row) * Kd + k0 + cc;
        f4v x0 = *(const f4v*)(src);
        f4v x1 = *(const f4v*)(src + 4);
        h8v t;
#pragma unroll
        for (int i = 0; i < 4; ++i) { t[i] = (_Float16)x0[i]; t[4 + i] = (_Float16)x1[i]; }
        *(h8v*)&As[row][cc] = t;
      } else {                       // f16 source: async DMA straight to LDS
        async_g2l_b128((const _Float16*)A + (size_t)(m0 + row) * Kd + k0 + cc,
                       &As[row][cc]);
      }
      async_g2l_b128(Wt + (size_t)(n0 + row) * Kd + k0 + cc, &Bs[row][cc]);
    }
    if (k0 + 32 < Kd) {  // prefetch next tiles (uniform branch)
      int c   = tid;
      int row = c >> 2;
      int cc  = (c & 3) * 8;
      __builtin_prefetch((const AT*)A + (size_t)(m0 + row) * Kd + k0 + 32 + cc, 0, 0);
      __builtin_prefetch(Wt + (size_t)(n0 + row) * Kd + k0 + 32 + cc, 0, 0);
    }
    wait_async<0>();
    __syncthreads();

    // ---- fragments (WMMA-native layouts) ----
    h16v af[2];
#pragma unroll
    for (int mt = 0; mt < 2; ++mt) {
      int r = wm * 32 + mt * 16 + l16;
      h8v x = *(const h8v*)&As[r][hf * 8];
      h8v y = *(const h8v*)&As[r][16 + hf * 8];
      af[mt] = cat8(x, y);
    }
    h16v bf[4];
#pragma unroll
    for (int nt = 0; nt < 4; ++nt) {
      int r = wn * 64 + nt * 16 + l16;
      h8v x = *(const h8v*)&Bs[r][hf * 8];
      h8v y = *(const h8v*)&Bs[r][16 + hf * 8];
      bf[nt] = cat8(x, y);
    }
#pragma unroll
    for (int mt = 0; mt < 2; ++mt)
#pragma unroll
      for (int nt = 0; nt < 4; ++nt) acc[mt][nt] = WMMA_F16(af[mt], bf[nt], acc[mt][nt]);
    __syncthreads();
  }

  // ---- epilogue: bias + store ----
#pragma unroll
  for (int nt = 0; nt < 4; ++nt) {
    int ncol = n0 + wn * 64 + nt * 16 + l16;
    float bv = bias[ncol];
#pragma unroll
    for (int mt = 0; mt < 2; ++mt) {
      int mb = m0 + wm * 32 + mt * 16 + 8 * hf;  // lane's 8 rows are mb..mb+7
      if (OM == OUT_F32_ROW) {
        float* o = (float*)outp;
#pragma unroll
        for (int r = 0; r < 8; ++r)
          o[(size_t)(mb + r) * N + ncol] = acc[mt][nt][r] + bv;
      } else if (OM == OUT_HEAD_ROW) {
        _Float16* o = (_Float16*)outp;
        int h = ncol >> 6, d = ncol & 63;
#pragma unroll
        for (int r = 0; r < 8; ++r) {
          int m = mb + r;
          int b = m / S, s = m - b * S;
          o[(((size_t)(b * 16 + h)) * S + s) * 64 + d] = (_Float16)(acc[mt][nt][r] + bv);
        }
      } else {  // OUT_HEAD_TRANS: [BH][64][S], 8 consecutive s -> one b128 store
        _Float16* o = (_Float16*)outp;
        int h = ncol >> 6, d = ncol & 63;
        int b = mb / S, s = mb - b * S;
        h8v t;
#pragma unroll
        for (int r = 0; r < 8; ++r) t[r] = (_Float16)(acc[mt][nt][r] + bv);
        *(h8v*)&o[(((size_t)(b * 16 + h)) * 64 + d) * S + s] = t;
      }
    }
  }
}

// --------------------------------------------------------------------------
// Flash attention, transposed-score formulation, double-buffered async K/V.
// Grid: (S/64, B*H), 128 threads (4 waves x 16 queries).
//   St[32kv x 16q] = Ktile @ Q^T   (A = K rows, B = Q rows, contract over d)
//   Ot[64d  x 16q] += Vt tile @ P  (A = Vt rows, B = P built in registers)
// Per-lane softmax stats (q = lane%16); one shfl_xor(16) reduction.
// Each thread issues 4 async b128 loads per KV tile; s_wait_asynccnt 4 in
// steady state keeps the next tile's DMA in flight during compute.
// --------------------------------------------------------------------------
__global__ __launch_bounds__(128) void flash_attn_wmma(
    const _Float16* __restrict__ Qp, const _Float16* __restrict__ Kp,
    const _Float16* __restrict__ Vpt, _Float16* __restrict__ attn, int S) {
  __shared__ _Float16 Ks[2][32][64];  // kv-local x d, double buffered
  __shared__ _Float16 Vs[2][64][32];  // d x kv-local, double buffered

  const int tid  = threadIdx.x;
  const int lane = tid & 31;
  const int w    = tid >> 5;
  const int l16  = lane & 15;
  const int hf   = lane >> 4;
  const int bh   = blockIdx.y;
  const int bb   = bh >> 4;
  const int hh   = bh & 15;
  const int q    = blockIdx.x * 64 + w * 16 + l16;  // this lane's query column

  const _Float16* Qb = Qp + (size_t)bh * S * 64;
  const _Float16* Kb = Kp + (size_t)bh * S * 64;
  const _Float16* Vb = Vpt + (size_t)bh * 64 * S;

  // Q row q as B-operand fragments for d-chunks [0,32) and [32,64)
  h16v qb[2];
#pragma unroll
  for (int dd = 0; dd < 2; ++dd) {
    const _Float16* p = Qb + (size_t)q * 64 + dd * 32 + hf * 8;
    qb[dd] = cat8(*(const h8v*)p, *(const h8v*)(p + 16));
  }

  // issue one KV tile's async loads (4 x b128 per thread, all in-bounds)
  auto issue_tile = [&](int kv0, int buf) {
#pragma unroll
    for (int it = 0; it < 2; ++it) {
      int c = tid + it * 128;  // 0..255
      {
        int row = c >> 3, cc = (c & 7) * 8;
        async_g2l_b128(Kb + (size_t)(kv0 + row) * 64 + cc, &Ks[buf][row][cc]);
      }
      {
        int row = c >> 2, cc = (c & 3) * 8;
        async_g2l_b128(Vb + (size_t)row * S + kv0 + cc, &Vs[buf][row][cc]);
      }
    }
  };

  const float scale = 0.125f;  // 1/sqrt(64)
  float m_prev = -1e30f, l = 0.f;
  f8v o[4];
#pragma unroll
  for (int t = 0; t < 4; ++t) o[t] = f8zero();

  const int nT = S / 32;
  issue_tile(0, 0);
  for (int i = 0; i < nT; ++i) {
    const int cur = i & 1;
    if (i + 1 < nT) {              // uniform branch
      issue_tile((i + 1) * 32, cur ^ 1);
      wait_async<4>();             // older 4 (this tile) done; next 4 in flight
    } else {
      wait_async<0>();
    }
    __syncthreads();               // all waves' current-tile DMA visible

    // ---- St = Ktile @ Q^T, accumulated over the two d-chunks ----
    f8v s0 = f8zero(), s1 = f8zero();
#pragma unroll
    for (int dd = 0; dd < 2; ++dd) {
      {
        int r = l16;  // kv 0..15
        h16v a = cat8(*(const h8v*)&Ks[cur][r][dd * 32 + hf * 8],
                      *(const h8v*)&Ks[cur][r][dd * 32 + 16 + hf * 8]);
        s0 = WMMA_F16(a, qb[dd], s0);
      }
      {
        int r = 16 + l16;  // kv 16..31
        h16v a = cat8(*(const h8v*)&Ks[cur][r][dd * 32 + hf * 8],
                      *(const h8v*)&Ks[cur][r][dd * 32 + 16 + hf * 8]);
        s1 = WMMA_F16(a, qb[dd], s1);
      }
    }

    // ---- online softmax (lane holds 16 of 32 kv scores; partner has rest) ----
    float sv[16];
#pragma unroll
    for (int r = 0; r < 8; ++r) { sv[r] = s0[r] * scale; sv[8 + r] = s1[r] * scale; }
    float mloc = sv[0];
#pragma unroll
    for (int i2 = 1; i2 < 16; ++i2) mloc = fmaxf(mloc, sv[i2]);
    mloc = fmaxf(mloc, __shfl_xor(mloc, 16, 32));
    float m_new = fmaxf(m_prev, mloc);
    float corr  = __expf(m_prev - m_new);

    float p[16], ssum = 0.f;
#pragma unroll
    for (int i2 = 0; i2 < 16; ++i2) { p[i2] = __expf(sv[i2] - m_new); ssum += p[i2]; }
    ssum += __shfl_xor(ssum, 16, 32);
    l = l * corr + ssum;
#pragma unroll
    for (int t = 0; t < 4; ++t)
#pragma unroll
      for (int r = 0; r < 8; ++r) o[t][r] *= corr;

    // ---- rebuild P as B-operand in registers (lane-pair shuffle only) ----
    // lane(hf=0) holds kv {r, 16+r}; needs kv {0..15}. lane(hf=1) needs {16..31}.
    h16v pb;
#pragma unroll
    for (int r = 0; r < 8; ++r) {
      float part0 = __shfl_xor(p[r], 16, 32);      // partner's kv r+8 / r
      float part1 = __shfl_xor(p[8 + r], 16, 32);  // partner's kv 24+r / 16+r
      pb[r]     = (_Float16)(hf ? part1 : p[r]);
      pb[8 + r] = (_Float16)(hf ? p[8 + r] : part0);
    }

    // ---- Ot += Vt tile @ P ----
#pragma unroll
    for (int t = 0; t < 4; ++t) {
      int r = t * 16 + l16;  // d row
      h16v a = cat8(*(const h8v*)&Vs[cur][r][hf * 16],
                    *(const h8v*)&Vs[cur][r][hf * 16 + 8]);
      o[t] = WMMA_F16(a, pb, o[t]);
    }
    __syncthreads();   // all reads of buf `cur` done before it is refilled
    m_prev = m_new;
  }

  // ---- normalize + store: attn[(b*S+q)][h*64 + d], 8 contiguous d per lane ----
  float inv = 1.f / l;
#pragma unroll
  for (int t = 0; t < 4; ++t) {
    h8v s;
#pragma unroll
    for (int r = 0; r < 8; ++r) s[r] = (_Float16)(o[t][r] * inv);
    *(h8v*)&attn[((size_t)(bb * S + q)) * 1024 + hh * 64 + t * 16 + hf * 8] = s;
  }
}

// --------------------------------------------------------------------------
extern "C" void kernel_launch(void* const* d_in, const int* in_sizes, int n_in,
                              void* d_out, int out_size, void* d_ws, size_t ws_size,
                              hipStream_t stream) {
  (void)in_sizes; (void)n_in; (void)out_size; (void)ws_size;
  const int B = 4, S = 2048, H = 16, N = 1024, QD = 1024, KD = 768;
  const int M = B * S;  // 8192

  const float* query = (const float*)d_in[0];
  const float* key   = (const float*)d_in[1];
  const float* value = (const float*)d_in[2];
  const float* Wq = (const float*)d_in[3]; const float* bq = (const float*)d_in[4];
  const float* Wk = (const float*)d_in[5]; const float* bk = (const float*)d_in[6];
  const float* Wv = (const float*)d_in[7]; const float* bv = (const float*)d_in[8];
  const float* Wo = (const float*)d_in[9]; const float* bo = (const float*)d_in[10];

  // workspace carve-up (f16 buffers), 256B aligned
  char* p = (char*)d_ws;
  auto carve = [&](size_t bytes) {
    void* r = (void*)p;
    p += (bytes + 255) & ~(size_t)255;
    return r;
  };
  _Float16* Wqt  = (_Float16*)carve((size_t)QD * N * 2);  // [N][QD]
  _Float16* Wkt  = (_Float16*)carve((size_t)KD * N * 2);  // [N][KD]
  _Float16* Wvt  = (_Float16*)carve((size_t)KD * N * 2);  // [N][KD]
  _Float16* Wot  = (_Float16*)carve((size_t)QD * N * 2);  // [N][QD]
  _Float16* Qp   = (_Float16*)carve((size_t)M * N * 2);   // [BH][S][64]
  _Float16* Kpj  = (_Float16*)carve((size_t)M * N * 2);   // [BH][S][64]
  _Float16* Vpt  = (_Float16*)carve((size_t)M * N * 2);   // [BH][64][S]
  _Float16* attn = (_Float16*)carve((size_t)M * N * 2);   // [B*S][1024]

  // 1) weight transposes
  {
    int tq = QD * N, tk = KD * N;
    wcvt_transpose<<<(tq + 255) / 256, 256, 0, stream>>>(Wq, Wqt, QD, N, tq);
    wcvt_transpose<<<(tk + 255) / 256, 256, 0, stream>>>(Wk, Wkt, KD, N, tk);
    wcvt_transpose<<<(tk + 255) / 256, 256, 0, stream>>>(Wv, Wvt, KD, N, tk);
    wcvt_transpose<<<(tq + 255) / 256, 256, 0, stream>>>(Wo, Wot, QD, N, tq);
  }

  dim3 gg(N / 128, M / 128);  // (8, 64)
  // 2) projections
  gemm_wmma_bias<float, OUT_HEAD_ROW><<<gg, 256, 0, stream>>>(query, Wqt, bq, Qp,  N, QD, S);
  gemm_wmma_bias<float, OUT_HEAD_ROW><<<gg, 256, 0, stream>>>(key,   Wkt, bk, Kpj, N, KD, S);
  gemm_wmma_bias<float, OUT_HEAD_TRANS><<<gg, 256, 0, stream>>>(value, Wvt, bv, Vpt, N, KD, S);
  // 3) attention
  flash_attn_wmma<<<dim3(S / 64, B * H), 128, 0, stream>>>(Qp, Kpj, Vpt, attn, S);
  // 4) output projection -> fp32 d_out
  gemm_wmma_bias<_Float16, OUT_F32_ROW><<<gg, 256, 0, stream>>>(attn, Wot, bo, (float*)d_out, N, QD, S);
}
